// GNNModule_9259949490279
// MI455X (gfx1250) — matrix operand order repaired and checked by reading, not compiled, not run on hardware
//
#include <hip/hip_runtime.h>
#include <hip/hip_bf16.h>
#include <math.h>

typedef __attribute__((ext_vector_type(16))) __bf16 v16bf;
typedef __attribute__((ext_vector_type(8)))  float  v8f;
typedef __attribute__((ext_vector_type(4)))  float  f32x4;

#define N_NODES 32768
#define N_EDGES 65536
#define BATCH   512
#define D       64
#define EH      32
#define KTOT    2112      // 2048 (z = x (x) h) + 64 (bias block carrying eb2)
#define NSLICE  66        // KTOT / 32
#define NTILES32 (N_EDGES / 32)
#define EPSV    1e-5f

// ---------------------------------------------------------------------------
// Generic fill
// ---------------------------------------------------------------------------
__global__ void fill_f32(float* p, float v, int n) {
    int t = blockIdx.x * blockDim.x + threadIdx.x;
    if (t < n) p[t] = v;
}

// ---------------------------------------------------------------------------
// Pre-pack B matrix (M2, KTOT x 64) into WMMA-fragment lane order, bf16.
// M2[i*EH+k, o] = ew2[(i*D+o)*EH + k]  for rows < 2048
// M2[2048+i, o] = eb2[i*D + o]         (bias block; A supplies x_src there)
// Layout: flat index = (((s*4 + nt)*32 + lane)*16 + elem)
//   B 32x16 bf16 frag: lanes 0-15 -> K 0..7,16..23 local, lanes 16-31 -> K 8..15,24..31,
//   elem e -> K = 8*(lane>=16) + e (+8 for e>=8); col = nt*16 + (lane&15).
// ---------------------------------------------------------------------------
__global__ void prepack_b(const float* __restrict__ ew2,
                          const float* __restrict__ eb2,
                          __bf16* __restrict__ packed) {
    int t = blockIdx.x * blockDim.x + threadIdx.x;
    const int total = NSLICE * 4 * 32 * 16;
    if (t >= total) return;
    int elem = t & 15;
    int lane = (t >> 4) & 31;
    int nt   = (t >> 9) & 3;
    int s    = t >> 11;
    int v = elem >> 1, half = elem & 1;
    int klocal = ((lane >= 16) ? 8 : 0) + 2 * v + half + ((elem >= 8) ? 8 : 0);
    int col = nt * 16 + (lane & 15);
    int row = s * 32 + klocal;
    float val;
    if (row < 2048) {
        int i = row >> 5, k = row & 31;
        val = ew2[(i * D + col) * EH + k];
    } else {
        int i = row - 2048;
        val = eb2[i * D + col];
    }
    packed[t] = (__bf16)val;
}

// ---------------------------------------------------------------------------
// Fused NNConv edge GEMM: one wave per 32-edge tile (2x M-blocked).
//   h[e,k]  = relu(ef[e]*ew1[k]+eb1[k])   (LDS, then hoisted to 16 regs/lane)
//   xs[e,i] = x_in[src[e], i]             (LDS)
//   K slices 0..63 : A[e,k] = xs[e,s] * h[e,k]   (outer product in regs)
//   K slices 64,65 : A[e,k] = xs[e,(s-64)*32+k]  (bias block, unrolled)
//   acc[2][16x64] += A @ Bpacked  via v_wmma_f32_16x16x32_bf16 (8 per slice,
//   B fragment shared by both M tiles)
// Epilogue: global f32 atomics scatter-add into sacc[dst], deg count.
// ---------------------------------------------------------------------------
__global__ __launch_bounds__(256) void edge_nnconv(
    const float* __restrict__ x_in, const float* __restrict__ edge_feat,
    const int* __restrict__ src, const int* __restrict__ dst,
    const float* __restrict__ ew1, const float* __restrict__ eb1,
    const __bf16* __restrict__ packed,
    float* __restrict__ sacc, float* __restrict__ deg) {
    __shared__ float h_lds[8][32][36];   // stride 36: banks spread, 16B aligned
    __shared__ float xs_lds[8][32][68];  // stride 68: banks spread, 16B aligned
    const int w    = threadIdx.x >> 5;
    const int lane = threadIdx.x & 31;
    const int tile = blockIdx.x * 8 + w;
    const int e0   = tile * 32;

    for (int t = lane; t < 32 * 32; t += 32) {
        int el = t >> 5, k = t & 31;
        float hv = edge_feat[e0 + el] * ew1[k] + eb1[k];
        h_lds[w][el][k] = hv > 0.f ? hv : 0.f;
    }
    for (int t = lane; t < 32 * 64; t += 32) {
        int el = t >> 6, i = t & 63;
        xs_lds[w][el][i] = x_in[(size_t)src[e0 + el] * D + i];
    }
    __syncthreads();

    const int mA = lane & 15;        // edges e0+0..15
    const int mB = 16 + (lane & 15); // edges e0+16..31
    const int kb = (lane >> 4) << 3; // hi lanes carry K+8

    // hoist h into element-ordered registers: elem e<8 -> K=kb+e, else kb+16+e-8
    f32x4 hA0 = *(const f32x4*)&h_lds[w][mA][kb];
    f32x4 hA1 = *(const f32x4*)&h_lds[w][mA][kb + 4];
    f32x4 hA2 = *(const f32x4*)&h_lds[w][mA][kb + 16];
    f32x4 hA3 = *(const f32x4*)&h_lds[w][mA][kb + 20];
    f32x4 hB0 = *(const f32x4*)&h_lds[w][mB][kb];
    f32x4 hB1 = *(const f32x4*)&h_lds[w][mB][kb + 4];
    f32x4 hB2 = *(const f32x4*)&h_lds[w][mB][kb + 16];
    f32x4 hB3 = *(const f32x4*)&h_lds[w][mB][kb + 20];
    float hA[16], hB[16];
#pragma unroll
    for (int j = 0; j < 4; ++j) {
        hA[j] = hA0[j]; hA[4 + j] = hA1[j]; hA[8 + j] = hA2[j]; hA[12 + j] = hA3[j];
        hB[j] = hB0[j]; hB[4 + j] = hB1[j]; hB[8 + j] = hB2[j]; hB[12 + j] = hB3[j];
    }

    v8f zero8 = {0.f, 0.f, 0.f, 0.f, 0.f, 0.f, 0.f, 0.f};
    v8f accA[4] = {zero8, zero8, zero8, zero8};
    v8f accB[4] = {zero8, zero8, zero8, zero8};

    // ---- main K loop: 64 outer-product slices, branch-free ----
    for (int s = 0; s < 64; s += 4) {
        f32x4 xsA = *(const f32x4*)&xs_lds[w][mA][s];
        f32x4 xsB = *(const f32x4*)&xs_lds[w][mB][s];
#pragma unroll
        for (int u = 0; u < 4; ++u) {
            float xvA = xsA[u], xvB = xsB[u];
            v16bf aA, aB;
#pragma unroll
            for (int e = 0; e < 16; ++e) {
                aA[e] = (__bf16)(xvA * hA[e]);
                aB[e] = (__bf16)(xvB * hB[e]);
            }
            const __bf16* bp = packed + ((size_t)((s + u) * 4) * 32 + lane) * 16;
#pragma unroll
            for (int nt = 0; nt < 4; ++nt) {
                v16bf b = *(const v16bf*)(bp + (size_t)nt * 32 * 16);
                accA[nt] = __builtin_amdgcn_wmma_f32_16x16x32_bf16(
                    false, aA, false, b, (short)0, accA[nt], false, false);
                accB[nt] = __builtin_amdgcn_wmma_f32_16x16x32_bf16(
                    false, aB, false, b, (short)0, accB[nt], false, false);
            }
        }
    }

    // ---- bias slices s = 64, 65: A[e,k] = xs[e, (s-64)*32 + k] ----
#pragma unroll
    for (int sb = 0; sb < 2; ++sb) {
        int off = sb * 32;
        f32x4 xA0 = *(const f32x4*)&xs_lds[w][mA][off + kb];
        f32x4 xA1 = *(const f32x4*)&xs_lds[w][mA][off + kb + 4];
        f32x4 xA2 = *(const f32x4*)&xs_lds[w][mA][off + kb + 16];
        f32x4 xA3 = *(const f32x4*)&xs_lds[w][mA][off + kb + 20];
        f32x4 xB0 = *(const f32x4*)&xs_lds[w][mB][off + kb];
        f32x4 xB1 = *(const f32x4*)&xs_lds[w][mB][off + kb + 4];
        f32x4 xB2 = *(const f32x4*)&xs_lds[w][mB][off + kb + 16];
        f32x4 xB3 = *(const f32x4*)&xs_lds[w][mB][off + kb + 20];
        v16bf aA, aB;
#pragma unroll
        for (int j = 0; j < 4; ++j) {
            aA[j]      = (__bf16)xA0[j]; aB[j]      = (__bf16)xB0[j];
            aA[4 + j]  = (__bf16)xA1[j]; aB[4 + j]  = (__bf16)xB1[j];
            aA[8 + j]  = (__bf16)xA2[j]; aB[8 + j]  = (__bf16)xB2[j];
            aA[12 + j] = (__bf16)xA3[j]; aB[12 + j] = (__bf16)xB3[j];
        }
        const __bf16* bp = packed + ((size_t)((64 + sb) * 4) * 32 + lane) * 16;
#pragma unroll
        for (int nt = 0; nt < 4; ++nt) {
            v16bf b = *(const v16bf*)(bp + (size_t)nt * 32 * 16);
            accA[nt] = __builtin_amdgcn_wmma_f32_16x16x32_bf16(
                false, aA, false, b, (short)0, accA[nt], false, false);
            accB[nt] = __builtin_amdgcn_wmma_f32_16x16x32_bf16(
                false, aB, false, b, (short)0, accB[nt], false, false);
        }
    }

    // C layout: lanes 0-15 N=lane M=r, lanes 16-31 N=lane-16 M=r+8
    const int N     = lane & 15;
    const int mbase = (lane >> 4) * 8;
#pragma unroll
    for (int r = 0; r < 8; ++r) {
        int dnA = dst[e0 + mbase + r];
        int dnB = dst[e0 + 16 + mbase + r];
        float* oA = sacc + (size_t)dnA * D;
        float* oB = sacc + (size_t)dnB * D;
#pragma unroll
        for (int nt = 0; nt < 4; ++nt) {
            atomicAdd(oA + nt * 16 + N, accA[nt][r]);
            atomicAdd(oB + nt * 16 + N, accB[nt][r]);
        }
    }
    atomicAdd(deg + dst[e0 + lane], 1.0f);
}

// scatter-mean + conv bias + relu
__global__ void node_update(const float* __restrict__ sacc,
                            const float* __restrict__ deg,
                            const float* __restrict__ cb,
                            float* __restrict__ x_out) {
    int t = blockIdx.x * blockDim.x + threadIdx.x;
    if (t >= N_NODES * D) return;
    int n = t >> 6, i = t & 63;
    float dg = deg[n]; dg = dg > 1.f ? dg : 1.f;
    float v = sacc[t] / dg + cb[i];
    x_out[t] = v > 0.f ? v : 0.f;
}

__global__ void layernorm_nodes(float* __restrict__ x,
                                const float* __restrict__ g,
                                const float* __restrict__ b) {
    int n = blockIdx.x * blockDim.x + threadIdx.x;
    if (n >= N_NODES) return;
    float v[D];
    float mu = 0.f;
#pragma unroll
    for (int i = 0; i < D; ++i) { v[i] = x[(size_t)n * D + i]; mu += v[i]; }
    mu *= (1.f / D);
    float var = 0.f;
#pragma unroll
    for (int i = 0; i < D; ++i) { float d = v[i] - mu; var += d * d; }
    var *= (1.f / D);
    float inv = rsqrtf(var + EPSV);
#pragma unroll
    for (int i = 0; i < D; ++i)
        x[(size_t)n * D + i] = (v[i] - mu) * inv * g[i] + b[i];
}

// segment boundaries from sorted node_graph_ids: seg[b] = lower_bound(b)
__global__ void seg_bounds(const int* __restrict__ gid, int* __restrict__ seg) {
    int b = blockIdx.x * blockDim.x + threadIdx.x;
    if (b > BATCH) return;
    int lo = 0, hi = N_NODES;
    while (lo < hi) {
        int mid = (lo + hi) >> 1;
        if (gid[mid] < b) lo = mid + 1; else hi = mid;
    }
    seg[b] = lo;
}

// ------------------------- Set2Set pieces ----------------------------------
__device__ __forceinline__ float sigm(float x) { return 1.f / (1.f + __expf(-x)); }

__global__ void lstm_gates(const float* __restrict__ inp, int in_dim,
                           const float* __restrict__ wih, const float* __restrict__ whh,
                           const float* __restrict__ bih, const float* __restrict__ bhh,
                           const float* __restrict__ hprev, float* __restrict__ gates) {
    int t = blockIdx.x * blockDim.x + threadIdx.x;
    if (t >= BATCH * 4 * D) return;
    int b = t >> 8, j = t & 255;
    float acc = bih[j] + bhh[j];
    const float* ir = inp + (size_t)b * in_dim;
    const float* wr = wih + (size_t)j * in_dim;
    for (int k = 0; k < in_dim; ++k) acc += ir[k] * wr[k];
    const float* hr = hprev + (size_t)b * D;
    const float* wh = whh + (size_t)j * D;
#pragma unroll 8
    for (int k = 0; k < D; ++k) acc += hr[k] * wh[k];
    gates[t] = acc;
}

__global__ void lstm_update(const float* __restrict__ gates,
                            float* __restrict__ c, float* __restrict__ h) {
    int t = blockIdx.x * blockDim.x + threadIdx.x;
    if (t >= BATCH * D) return;
    int b = t >> 6, d = t & 63;
    const float* g = gates + (size_t)b * 256;
    float ig = sigm(g[d]), fg = sigm(g[64 + d]);
    float gg = tanhf(g[128 + d]), og = sigm(g[192 + d]);
    float cn = fg * c[t] + ig * gg;
    c[t] = cn;
    h[t] = og * tanhf(cn);
}

__global__ void attn_e(const float* __restrict__ x, const float* __restrict__ q,
                       const int* __restrict__ gid, float* __restrict__ ebuf) {
    int n = blockIdx.x * blockDim.x + threadIdx.x;
    if (n >= N_NODES) return;
    const float* xr = x + (size_t)n * D;
    const float* qr = q + (size_t)gid[n] * D;
    float s = 0.f;
#pragma unroll 8
    for (int i = 0; i < D; ++i) s += xr[i] * qr[i];
    ebuf[n] = s;
}

// one wave (32 lanes) per graph: segment max then sum of exp
__global__ __launch_bounds__(256) void seg_softmax(
    const float* __restrict__ ebuf, const int* __restrict__ seg,
    float* __restrict__ mx, float* __restrict__ zz) {
    int w = (blockIdx.x * blockDim.x + threadIdx.x) >> 5;
    int lane = threadIdx.x & 31;
    if (w >= BATCH) return;
    int s0 = seg[w], s1 = seg[w + 1];
    float m = -1e30f;
    for (int n = s0 + lane; n < s1; n += 32) m = fmaxf(m, ebuf[n]);
    for (int off = 16; off; off >>= 1) m = fmaxf(m, __shfl_xor(m, off, 32));
    float z = 0.f;
    for (int n = s0 + lane; n < s1; n += 32) z += __expf(ebuf[n] - m);
    for (int off = 16; off; off >>= 1) z += __shfl_xor(z, off, 32);
    if (lane == 0) { mx[w] = m; zz[w] = z > 0.f ? z : 1.f; }
}

// block (64 threads) per graph: alpha-weighted readout, build q_star
__global__ __launch_bounds__(64) void readout_qstar(
    const float* __restrict__ x, const float* __restrict__ ebuf,
    const int* __restrict__ seg, const float* __restrict__ mx,
    const float* __restrict__ zz, const float* __restrict__ h2,
    float* __restrict__ qstar) {
    int b = blockIdx.x, d = threadIdx.x;
    int s0 = seg[b], s1 = seg[b + 1];
    float m = mx[b], zi = 1.f / zz[b];
    float r = 0.f;
    for (int n = s0; n < s1; ++n)
        r += __expf(ebuf[n] - m) * zi * x[(size_t)n * D + d];
    qstar[(size_t)b * 2 * D + d]     = h2[(size_t)b * D + d];
    qstar[(size_t)b * 2 * D + D + d] = r;
}

// ----------------- fused head: LN -> fc1 -> BN -> relu -> fc2 -> relu ------
__global__ __launch_bounds__(128) void head(
    const float* __restrict__ qstar,
    const float* __restrict__ gng, const float* __restrict__ gnb,
    const float* __restrict__ fc1w, const float* __restrict__ fc1b,
    const float* __restrict__ bng, const float* __restrict__ bnb,
    const float* __restrict__ bnm, const float* __restrict__ bnv,
    const float* __restrict__ fc2w, const float* __restrict__ fc2b,
    float* __restrict__ out) {
    __shared__ float red[128];
    __shared__ float gsh[128];
    __shared__ float ysh[128];
    int b = blockIdx.x, j = threadIdx.x;
    float v = qstar[(size_t)b * 128 + j];
    red[j] = v; __syncthreads();
    for (int st = 64; st > 0; st >>= 1) {
        if (j < st) red[j] += red[j + st];
        __syncthreads();
    }
    float mu = red[0] * (1.f / 128.f);
    __syncthreads();
    float dv = v - mu;
    red[j] = dv * dv; __syncthreads();
    for (int st = 64; st > 0; st >>= 1) {
        if (j < st) red[j] += red[j + st];
        __syncthreads();
    }
    float var = red[0] * (1.f / 128.f);
    __syncthreads();
    gsh[j] = dv * rsqrtf(var + EPSV) * gng[j] + gnb[j];
    __syncthreads();
    float acc = fc1b[j];
    const float* wr = fc1w + (size_t)j * 128;
#pragma unroll 8
    for (int k = 0; k < 128; ++k) acc += gsh[k] * wr[k];
    acc = (acc - bnm[j]) * rsqrtf(bnv[j] + EPSV) * bng[j] + bnb[j];
    ysh[j] = acc > 0.f ? acc : 0.f;
    __syncthreads();
    if (j < D) {
        float o = fc2b[j];
        const float* w2 = fc2w + (size_t)j * 128;
#pragma unroll 8
        for (int k = 0; k < 128; ++k) o += ysh[k] * w2[k];
        out[(size_t)b * D + j] = o > 0.f ? o : 0.f;
    }
}

// ---------------------------------------------------------------------------
extern "C" void kernel_launch(void* const* d_in, const int* in_sizes, int n_in,
                              void* d_out, int out_size, void* d_ws, size_t ws_size,
                              hipStream_t stream) {
    const float* atom  = (const float*)d_in[0];
    const float* efeat = (const float*)d_in[1];
    const int*   srcI  = (const int*)d_in[2];
    const int*   dstI  = (const int*)d_in[3];
    const int*   gid   = (const int*)d_in[4];
    const float* ew1_[2] = {(const float*)d_in[5],  (const float*)d_in[10]};
    const float* eb1_[2] = {(const float*)d_in[6],  (const float*)d_in[11]};
    const float* ew2_[2] = {(const float*)d_in[7],  (const float*)d_in[12]};
    const float* eb2_[2] = {(const float*)d_in[8],  (const float*)d_in[13]};
    const float* cb_[2]  = {(const float*)d_in[9],  (const float*)d_in[14]};
    const float* ln_g = (const float*)d_in[15];
    const float* ln_b = (const float*)d_in[16];
    const float* wih[3] = {(const float*)d_in[17], (const float*)d_in[21], (const float*)d_in[25]};
    const float* whh[3] = {(const float*)d_in[18], (const float*)d_in[22], (const float*)d_in[26]};
    const float* bih[3] = {(const float*)d_in[19], (const float*)d_in[23], (const float*)d_in[27]};
    const float* bhh[3] = {(const float*)d_in[20], (const float*)d_in[24], (const float*)d_in[28]};
    const float* gng = (const float*)d_in[29];
    const float* gnb = (const float*)d_in[30];
    const float* fc1w = (const float*)d_in[31];
    const float* fc1b = (const float*)d_in[32];
    const float* bng = (const float*)d_in[33];
    const float* bnb = (const float*)d_in[34];
    const float* bnm = (const float*)d_in[35];
    const float* bnv = (const float*)d_in[36];
    const float* fc2w = (const float*)d_in[37];
    const float* fc2b = (const float*)d_in[38];

    // -------- workspace bump allocation (256B aligned) --------
    char* base = (char*)d_ws;
    size_t off = 0;
    auto alloc = [&](size_t bytes) {
        char* p = base + off;
        off += (bytes + 255) & ~size_t(255);
        return p;
    };
    float*  xA     = (float*)alloc((size_t)N_NODES * D * 4);
    float*  sacc   = (float*)alloc((size_t)N_NODES * D * 4);
    __bf16* packed = (__bf16*)alloc((size_t)NSLICE * 4 * 32 * 16 * 2);
    float*  deg    = (float*)alloc((size_t)N_NODES * 4);
    float*  ebuf   = (float*)alloc((size_t)N_NODES * 4);
    int*    seg    = (int*)alloc((size_t)(BATCH + 1) * 4);
    float*  mx     = (float*)alloc((size_t)BATCH * 4);
    float*  zz     = (float*)alloc((size_t)BATCH * 4);
    float*  qstar  = (float*)alloc((size_t)BATCH * 2 * D * 4);
    float*  hst    = (float*)alloc((size_t)3 * BATCH * D * 4);
    float*  cst    = (float*)alloc((size_t)3 * BATCH * D * 4);
    float*  gates  = (float*)alloc((size_t)BATCH * 4 * D * 4);

    const int T = 256;
    auto blks = [](int n, int t) { return (n + t - 1) / t; };

    // -------- 2x NNConv layers --------
    for (int l = 0; l < 2; ++l) {
        const float* x_in = (l == 0) ? atom : xA;
        const int npk = NSLICE * 4 * 32 * 16;
        prepack_b<<<blks(npk, T), T, 0, stream>>>(ew2_[l], eb2_[l], packed);
        fill_f32<<<blks(N_NODES * D, T), T, 0, stream>>>(sacc, 0.f, N_NODES * D);
        fill_f32<<<blks(N_NODES, T), T, 0, stream>>>(deg, 0.f, N_NODES);
        edge_nnconv<<<NTILES32 / 8, T, 0, stream>>>(x_in, efeat, srcI, dstI,
                                                    ew1_[l], eb1_[l], packed, sacc, deg);
        node_update<<<blks(N_NODES * D, T), T, 0, stream>>>(sacc, deg, cb_[l], xA);
    }
    layernorm_nodes<<<blks(N_NODES, T), T, 0, stream>>>(xA, ln_g, ln_b);

    // -------- Set2Set --------
    seg_bounds<<<blks(BATCH + 1, T), T, 0, stream>>>(gid, seg);
    // zero q_star, h, c (laid out contiguously)
    fill_f32<<<blks(BATCH * 2 * D + 6 * BATCH * D, T), T, 0, stream>>>(
        qstar, 0.f, BATCH * 2 * D + 6 * BATCH * D);

    float* h2 = hst + 2 * BATCH * D;
    for (int it = 0; it < 6; ++it) {
        const float* inp = qstar;
        int in_dim = 2 * D;
        for (int l = 0; l < 3; ++l) {
            float* hl = hst + (size_t)l * BATCH * D;
            float* cl = cst + (size_t)l * BATCH * D;
            lstm_gates<<<blks(BATCH * 4 * D, T), T, 0, stream>>>(
                inp, in_dim, wih[l], whh[l], bih[l], bhh[l], hl, gates);
            lstm_update<<<blks(BATCH * D, T), T, 0, stream>>>(gates, cl, hl);
            inp = hl;
            in_dim = D;
        }
        attn_e<<<blks(N_NODES, T), T, 0, stream>>>(xA, h2, gid, ebuf);
        seg_softmax<<<blks(BATCH * 32, T), T, 0, stream>>>(ebuf, seg, mx, zz);
        readout_qstar<<<BATCH, 64, 0, stream>>>(xA, ebuf, seg, mx, zz, h2, qstar);
    }

    // -------- head --------
    head<<<BATCH, 128, 0, stream>>>(qstar, gng, gnb, fc1w, fc1b,
                                    bng, bnb, bnm, bnv, fc2w, fc2b, (float*)d_out);
}
